// MS_Block_Conv_936302870978
// MI455X (gfx1250) — matrix operand order repaired and checked by reading, not compiled, not run on hardware
//
#include <hip/hip_runtime.h>
#include <hip/hip_bf16.h>

// ---------------------------------------------------------------------------
// Types for CDNA5 WMMA (wave32): v_wmma_f32_16x16x32_bf16
// ---------------------------------------------------------------------------
typedef __attribute__((ext_vector_type(16))) __bf16 v16bf;
typedef __attribute__((ext_vector_type(8)))  float  v8f;

union FragBF16 {
    v16bf v;
    uint4 q[2];   // 32 bytes = 16 bf16
};

__device__ __forceinline__ unsigned short f2bf(float f) {
    unsigned int u = __float_as_uint(f);
    unsigned int r = u + 0x7FFFu + ((u >> 16) & 1u);   // round-to-nearest-even
    return (unsigned short)(r >> 16);
}

// ---------------------------------------------------------------------------
// Direct global->LDS async copy (CDNA5 GLOBAL_LOAD_ASYNC_TO_LDS_*, ASYNCcnt).
// Probe-derived signatures (from clang-22 diagnostics):
//   b128: (v4i_as1*, v4i_as3*, Imm int offset, Imm int cpol)
//   b32 : (int_as1*, int_as3*, Imm int offset, Imm int cpol)
// where the vector type is GCC vector_size(16) int, AS1=global, AS3=LDS.
// ---------------------------------------------------------------------------
#if defined(__AMDGCN__) && \
    __has_builtin(__builtin_amdgcn_global_load_async_to_lds_b128) && \
    __has_builtin(__builtin_amdgcn_global_load_async_to_lds_b32)  && \
    __has_builtin(__builtin_amdgcn_s_wait_asynccnt)
#define USE_ASYNC_G2L 1
#else
#define USE_ASYNC_G2L 0
#endif

#if USE_ASYNC_G2L
typedef int v4i_vs __attribute__((vector_size(16)));
typedef __attribute__((address_space(1))) v4i_vs* gptr_v4i;
typedef __attribute__((address_space(3))) v4i_vs* lptr_v4i;
typedef __attribute__((address_space(1))) int*    gptr_i;
typedef __attribute__((address_space(3))) int*    lptr_i;
#endif

__device__ __forceinline__ void copy16_g2l(const unsigned short* g, unsigned short* l) {
#if USE_ASYNC_G2L
    __builtin_amdgcn_global_load_async_to_lds_b128(
        (gptr_v4i)(v4i_vs*)(void*)g, (lptr_v4i)(v4i_vs*)(void*)l, 0, 0);
#else
    *(uint4*)l = *(const uint4*)g;
#endif
}

__device__ __forceinline__ void copy4_g2l(const float* g, float* l) {
#if USE_ASYNC_G2L
    __builtin_amdgcn_global_load_async_to_lds_b32(
        (gptr_i)(int*)(void*)g, (lptr_i)(int*)(void*)l, 0, 0);
#else
    *l = *g;
#endif
}

__device__ __forceinline__ void wait_g2l() {
#if USE_ASYNC_G2L
    __builtin_amdgcn_s_wait_asynccnt(0);
#endif
}

// ---------------------------------------------------------------------------
// fp32 -> bf16 weight conversion
// ---------------------------------------------------------------------------
__global__ __launch_bounds__(256)
void cvt_bf16_kernel(const float* __restrict__ in, unsigned short* __restrict__ out, int n) {
    int i = blockIdx.x * 256 + threadIdx.x;
    if (i < n) out[i] = f2bf(in[i]);
}

// ---------------------------------------------------------------------------
// LayerNorm over C (row per block), writes bf16 A-matrix for GEMM1
// ---------------------------------------------------------------------------
__global__ __launch_bounds__(256)
void layernorm_kernel(const float* __restrict__ x, const float* __restrict__ g,
                      const float* __restrict__ b, unsigned short* __restrict__ out, int C) {
    __shared__ float rs[256], rs2[256];
    int row = blockIdx.x;
    int tid = threadIdx.x;
    const float* xr = x + (size_t)row * C;
    float s = 0.f, s2 = 0.f;
    for (int c = tid; c < C; c += 256) { float v = xr[c]; s += v; s2 += v * v; }
    rs[tid] = s; rs2[tid] = s2;
    __syncthreads();
    for (int o = 128; o > 0; o >>= 1) {
        if (tid < o) { rs[tid] += rs[tid + o]; rs2[tid] += rs2[tid + o]; }
        __syncthreads();
    }
    float mu  = rs[0] / (float)C;
    float var = rs2[0] / (float)C - mu * mu;
    float inv = rsqrtf(var + 1e-5f);
    for (int c = tid; c < C; c += 256) {
        float y = (xr[c] - mu) * inv * g[c] + b[c];
        out[(size_t)row * C + c] = f2bf(y);
    }
}

// ---------------------------------------------------------------------------
// bf16 WMMA GEMM:  Cout[m][n] = bn( sum_k A[m][k]*W[n][k] + bias[n] )
// Block tile 128x128, BK=32, 256 threads = 8 waves, each wave 64x32 (4x2 WMMA).
// A-frag: lane l (m=l%16, half=l/16): K halves {half*8..+7} and {16+half*8..+7}
// B-frag: lane l (n=l%16): K halves {half*8..+7} and {16+half*8..+7}  (W rows are K-contig)
// Tiles staged to LDS via async global->LDS copies when available.
// ---------------------------------------------------------------------------
#define BM 128
#define BN 128
#define BK 32

__global__ __launch_bounds__(256)
void gemm_bf16_bn(const unsigned short* __restrict__ A,   // [M][K] bf16
                  const unsigned short* __restrict__ W,   // [N][K] bf16
                  const float* __restrict__ bias,         // [N]
                  const float* __restrict__ bnp,          // [4][N]: g,b,m,v
                  float* __restrict__ Cout,               // [M][N]
                  int M, int N, int K, int nBlocksN) {
    __shared__ unsigned short lA[BM * BK];
    __shared__ unsigned short lB[BN * BK];

    const int bm   = blockIdx.x / nBlocksN;
    const int bn   = blockIdx.x % nBlocksN;
    const int tid  = threadIdx.x;
    const int wave = tid >> 5;       // 0..7 (wave32)
    const int lane = tid & 31;
    const int waveM = wave >> 2;     // 0..1 -> 64 rows each
    const int waveN = wave & 3;      // 0..3 -> 32 cols each

    v8f acc[4][2];
#pragma unroll
    for (int i = 0; i < 4; ++i)
#pragma unroll
        for (int j = 0; j < 2; ++j)
#pragma unroll
            for (int e = 0; e < 8; ++e) acc[i][j][e] = 0.f;

    const int rowA = bm * BM;
    const int rowW = bn * BN;
    const int mloc = lane & 15;
    const int half = lane >> 4;
    const int kb   = half * 16;

    for (int k0 = 0; k0 < K; k0 += BK) {
        // stage A & W tiles: 4096 bf16 each, 16B chunks, 2 per thread per tile
#pragma unroll
        for (int p = 0; p < 2; ++p) {
            int chunk = tid + p * 256;        // 0..511
            int r  = chunk >> 2;              // row within tile (0..127)
            int cc = (chunk & 3) * 8;         // bf16 offset within row
            copy16_g2l(&A[(size_t)(rowA + r) * K + k0 + cc], &lA[r * BK + cc]);
            copy16_g2l(&W[(size_t)(rowW + r) * K + k0 + cc], &lB[r * BK + cc]);
        }
        wait_g2l();
        __syncthreads();

        FragBF16 af[4], bfm[2];
#pragma unroll
        for (int i = 0; i < 4; ++i) {
            int r = waveM * 64 + i * 16 + mloc;
            af[i].q[0] = *(const uint4*)&lA[r * BK + half * 8];
            af[i].q[1] = *(const uint4*)&lA[r * BK + 16 + half * 8];
        }
#pragma unroll
        for (int j = 0; j < 2; ++j) {
            int c = waveN * 32 + j * 16 + mloc;
            bfm[j].q[0] = *(const uint4*)&lB[c * BK + half * 8];
            bfm[j].q[1] = *(const uint4*)&lB[c * BK + 16 + half * 8];
        }
#pragma unroll
        for (int i = 0; i < 4; ++i)
#pragma unroll
            for (int j = 0; j < 2; ++j)
                acc[i][j] = __builtin_amdgcn_wmma_f32_16x16x32_bf16(
                    false, af[i].v, false, bfm[j].v, (short)0, acc[i][j], false, false);
        __syncthreads();
    }

    // epilogue: bias + BatchNorm, fp32 store
    const int mh = half * 8;   // C/D layout: element e -> m = 8*(lane/16)+e, n = lane%16
#pragma unroll
    for (int j = 0; j < 2; ++j) {
        int col = bn * BN + waveN * 32 + j * 16 + mloc;
        float bb = bias[col];
        float gg = bnp[col], be = bnp[N + col], mm = bnp[2 * N + col], vv = bnp[3 * N + col];
        float scale = gg * rsqrtf(vv + 1e-5f);
#pragma unroll
        for (int i = 0; i < 4; ++i) {
            int rbase = bm * BM + waveM * 64 + i * 16 + mh;
#pragma unroll
            for (int e = 0; e < 8; ++e) {
                float y = (acc[i][j][e] + bb - mm) * scale + be;
                Cout[(size_t)(rbase + e) * N + col] = y;
            }
        }
    }
}

// ---------------------------------------------------------------------------
// LIF scan over T (input already BN'ed); fp32 spikes out. decay = 0.5, vth = 1
// ---------------------------------------------------------------------------
__global__ __launch_bounds__(256)
void lif_f32_kernel(const float* __restrict__ in, float* __restrict__ out, int BC, int T) {
    int idx = blockIdx.x * 256 + threadIdx.x;
    float v = 0.f;
    for (int t = 0; t < T; ++t) {
        size_t o = (size_t)t * BC + idx;
        v = v * 0.5f + in[o];
        float s = (v >= 1.0f) ? 1.0f : 0.0f;
        v -= v * s;
        out[o] = s;
    }
}

// ---------------------------------------------------------------------------
// Depthwise time-conv (K=31, SAME) + bias + BN2 + LIF, spikes stored as bf16.
// One block = 64 channels of one batch element; full time series in LDS.
// ---------------------------------------------------------------------------
__global__ __launch_bounds__(64)
void dw_bn_lif_kernel(const float* __restrict__ s1, const float* __restrict__ dw_w,
                      const float* __restrict__ dw_b, const float* __restrict__ bnp,
                      unsigned short* __restrict__ s2, int B, int C) {
    const int T = 128, KW = 31, PAD = 15, CH = 64;
    __shared__ float lds[(T + 2 * PAD) * CH];   // 158*64*4 = 40448 B

    int nCb = C / CH;
    int b   = blockIdx.x / nCb;
    int cb  = blockIdx.x % nCb;
    int tid = threadIdx.x;
    int c   = cb * CH + tid;
    size_t base = (size_t)b * C + c;

#pragma unroll
    for (int t = 0; t < PAD; ++t) {
        lds[t * CH + tid] = 0.f;
        lds[(T + PAD + t) * CH + tid] = 0.f;
    }
    for (int t = 0; t < T; ++t)
        copy4_g2l(&s1[(size_t)t * B * C + base], &lds[(t + PAD) * CH + tid]);
    wait_g2l();
    __syncthreads();

    float w[KW];
#pragma unroll
    for (int k = 0; k < KW; ++k) w[k] = dw_w[(size_t)c * KW + k];
    float bias = dw_b[c];
    float gg = bnp[c], be = bnp[C + c], mm = bnp[2 * C + c], vv = bnp[3 * C + c];
    float scale = gg * rsqrtf(vv + 1e-5f);

    float v = 0.f;
    for (int t = 0; t < T; ++t) {
        float acc = bias;
#pragma unroll
        for (int k = 0; k < KW; ++k) acc += w[k] * lds[(t + k) * CH + tid];
        float u = (acc - mm) * scale + be;
        v = v * 0.5f + u;
        float s = (v >= 1.0f) ? 1.0f : 0.0f;
        v -= v * s;
        s2[(size_t)t * B * C + base] = (s > 0.5f) ? (unsigned short)0x3F80 : (unsigned short)0;
    }
}

// ---------------------------------------------------------------------------
// Gate LIF: reads gate half of h3 (already BN3'ed), writes bf16 spikes
// ---------------------------------------------------------------------------
__global__ __launch_bounds__(256)
void gate_lif_kernel(const float* __restrict__ h3, unsigned short* __restrict__ g,
                     int B, int C, int T) {
    int idx = blockIdx.x * 256 + threadIdx.x;   // b*C + c
    int b = idx / C, c = idx % C;
    float v = 0.f;
    for (int t = 0; t < T; ++t) {
        size_t m = (size_t)t * B + b;
        float u = h3[m * (2 * C) + C + c];
        v = v * 0.5f + u;
        float s = (v >= 1.0f) ? 1.0f : 0.0f;
        v -= v * s;
        g[m * C + c] = (s > 0.5f) ? (unsigned short)0x3F80 : (unsigned short)0;
    }
}

// ---------------------------------------------------------------------------
// Final: gate2 = LIF(h4); h = out_half(h3) * gate2; out = LIF(h)
// ---------------------------------------------------------------------------
__global__ __launch_bounds__(256)
void final_kernel(const float* __restrict__ h3, const float* __restrict__ h4,
                  float* __restrict__ out, int B, int C, int T) {
    int idx = blockIdx.x * 256 + threadIdx.x;   // b*C + c
    int b = idx / C, c = idx % C;
    float vg = 0.f, vo = 0.f;
    for (int t = 0; t < T; ++t) {
        size_t m = (size_t)t * B + b;
        float u = h4[m * C + c];
        vg = vg * 0.5f + u;
        float sg = (vg >= 1.0f) ? 1.0f : 0.0f;
        vg -= vg * sg;
        float h = h3[m * (2 * C) + c] * sg;
        vo = vo * 0.5f + h;
        float s = (vo >= 1.0f) ? 1.0f : 0.0f;
        vo -= vo * s;
        out[m * C + c] = s;
    }
}

// ---------------------------------------------------------------------------
// Launch
// ---------------------------------------------------------------------------
extern "C" void kernel_launch(void* const* d_in, const int* in_sizes, int n_in,
                              void* d_out, int out_size, void* d_ws, size_t ws_size,
                              hipStream_t stream) {
    const int T = 128, B = 64, C = 768, C2 = 2 * C, M = T * B;

    const float* x        = (const float*)d_in[0];
    const float* ln_g     = (const float*)d_in[1];
    const float* ln_b     = (const float*)d_in[2];
    const float* pw1_w    = (const float*)d_in[3];
    const float* pw1_b    = (const float*)d_in[4];
    const float* bn1_p    = (const float*)d_in[5];
    const float* dw_w     = (const float*)d_in[6];
    const float* dw_b     = (const float*)d_in[7];
    const float* bn2_p    = (const float*)d_in[8];
    const float* pw2_w    = (const float*)d_in[9];
    const float* pw2_b    = (const float*)d_in[10];
    const float* bn3_p    = (const float*)d_in[11];
    const float* gsu_w    = (const float*)d_in[12];
    const float* gsu_b    = (const float*)d_in[13];
    const float* gsu_bn_p = (const float*)d_in[14];

    char* ws = (char*)d_ws;
    size_t off = 0;
    auto alloc = [&](size_t bytes) -> char* {
        char* p = ws + off;
        off += (bytes + 255) & ~(size_t)255;
        return p;
    };
    unsigned short* W1   = (unsigned short*)alloc((size_t)C * C * 2);
    unsigned short* W2   = (unsigned short*)alloc((size_t)C2 * C * 2);
    unsigned short* W3   = (unsigned short*)alloc((size_t)C * C * 2);
    unsigned short* Abuf = (unsigned short*)alloc((size_t)M * C * 2);  // h0 -> s2 -> gate spikes
    float* h1 = (float*)alloc((size_t)M * C * 4);                      // h1 -> h4
    float* s1 = (float*)alloc((size_t)M * C * 4);
    float* h3 = (float*)alloc((size_t)M * C2 * 4);

    // weights fp32 -> bf16
    cvt_bf16_kernel<<<(C * C + 255) / 256, 256, 0, stream>>>(pw1_w, W1, C * C);
    cvt_bf16_kernel<<<(C2 * C + 255) / 256, 256, 0, stream>>>(pw2_w, W2, C2 * C);
    cvt_bf16_kernel<<<(C * C + 255) / 256, 256, 0, stream>>>(gsu_w, W3, C * C);

    // layernorm -> bf16 A
    layernorm_kernel<<<M, 256, 0, stream>>>(x, ln_g, ln_b, Abuf, C);

    // GEMM1 + bias + BN1 -> h1
    gemm_bf16_bn<<<(M / BM) * (C / BN), 256, 0, stream>>>(
        Abuf, W1, pw1_b, bn1_p, h1, M, C, C, C / BN);

    // LIF1 -> s1 (fp32 spikes)
    lif_f32_kernel<<<(B * C) / 256, 256, 0, stream>>>(h1, s1, B * C, T);

    // depthwise + BN2 + LIF2 -> bf16 spikes (Abuf)
    dw_bn_lif_kernel<<<B * (C / 64), 64, 0, stream>>>(s1, dw_w, dw_b, bn2_p, Abuf, B, C);

    // GEMM2 + bias + BN3 -> h3 [M][2C]
    gemm_bf16_bn<<<(M / BM) * (C2 / BN), 256, 0, stream>>>(
        Abuf, W2, pw2_b, bn3_p, h3, M, C2, C, C2 / BN);

    // gate LIF -> bf16 spikes (Abuf)
    gate_lif_kernel<<<(B * C) / 256, 256, 0, stream>>>(h3, Abuf, B, C, T);

    // GSU GEMM + bias + gsu BN -> h4 (reuse h1)
    gemm_bf16_bn<<<(M / BM) * (C / BN), 256, 0, stream>>>(
        Abuf, W3, gsu_b, gsu_bn_p, h1, M, C, C, C / BN);

    // gate2 LIF, multiply with out-half, final LIF -> d_out
    final_kernel<<<(B * C) / 256, 256, 0, stream>>>(h3, h1, (float*)d_out, B, C, T);
}
